// RefactAttention_64673617543347
// MI455X (gfx1250) — matrix operand order repaired and checked by reading, not compiled, not run on hardware
//
#include <hip/hip_runtime.h>
#include <cmath>

// CDNA5 / gfx1250, wave32.
// GEMMs + attention matmuls: split-bf16 (bf16x3) on v_wmma_f32_16x16x32_bf16.
// Softmax statistics and accumulation stay in exact fp32.

typedef float v2f  __attribute__((ext_vector_type(2)));
typedef float v8f  __attribute__((ext_vector_type(8)));
typedef __bf16 v8bf  __attribute__((ext_vector_type(8)));
typedef __bf16 v16bf __attribute__((ext_vector_type(16)));

#define NUM_HEADS 32
#define HEAD_DIM  128
#define SEQ       2048
#define BATCH     2
#define EMB       4096
#define QKV_N     (EMB + 2 * HEAD_DIM)   // 4352

static __device__ __forceinline__ v8f wmma32(v16bf a, v16bf b, v8f c) {
  return __builtin_amdgcn_wmma_f32_16x16x32_bf16(false, a, false, b, (short)0,
                                                 c, false, false);
}

// split fp32 -> bf16 hi (truncated) + bf16 lo (residual); a ~= hi + lo
static __device__ __forceinline__ void splitbf(float a, unsigned short& h,
                                               unsigned short& l) {
  const unsigned int bits = __float_as_uint(a);
  const unsigned int hb   = bits & 0xFFFF0000u;
  h = (unsigned short)(hb >> 16);
  const float r = a - __uint_as_float(hb);
  l = (unsigned short)(__float_as_uint(r) >> 16);
}

static __device__ __forceinline__ __bf16 us2bf(unsigned short u) {
  return __builtin_bit_cast(__bf16, u);
}

static __device__ __forceinline__ v16bf cat8(v8bf a, v8bf b) {
  return __builtin_shufflevector(a, b, 0, 1, 2, 3, 4, 5, 6, 7, 8, 9, 10, 11,
                                 12, 13, 14, 15);
}

// split 8 consecutive floats (16B-aligned) into bf16 hi/lo vectors
static __device__ __forceinline__ void split8(const float* src, v8bf& h8,
                                              v8bf& l8) {
  const float4 a = *(const float4*)(src);
  const float4 b = *(const float4*)(src + 4);
  const float vals[8] = {a.x, a.y, a.z, a.w, b.x, b.y, b.z, b.w};
#pragma unroll
  for (int i = 0; i < 8; ++i) {
    unsigned short h, l;
    splitbf(vals[i], h, l);
    h8[i] = us2bf(h);
    l8[i] = us2bf(l);
  }
}

// ---------------------------------------------------------------------------
// C[M,N] = A[M,K] @ B[K,N], all fp32 row-major, computed via bf16x3.
// 256 threads = 8 waves; block tile 128x128; K-block 64; wave tile 64x32.
// ---------------------------------------------------------------------------
__global__ __launch_bounds__(256) void gemm_bf16x3(
    const float* __restrict__ A, const float* __restrict__ B,
    float* __restrict__ C, int M, int N, int K) {
  constexpr int LDT = 72;
  __shared__ unsigned short As_hi[128 * LDT];
  __shared__ unsigned short As_lo[128 * LDT];
  __shared__ unsigned short Bs_hi[128 * LDT];
  __shared__ unsigned short Bs_lo[128 * LDT];

  const int tid  = threadIdx.x;
  const int lane = tid & 31;
  const int wave = tid >> 5;
  const int ln   = lane & 15;
  const int g    = lane >> 4;
  const int wm   = (wave >> 2) * 64;   // wave grid 2 (M) x 4 (N)
  const int wn   = (wave & 3) * 32;
  const int mb   = blockIdx.y * 128;
  const int nb   = blockIdx.x * 128;

  v8f acc[4][2];
#pragma unroll
  for (int i = 0; i < 4; ++i)
#pragma unroll
    for (int j = 0; j < 2; ++j) acc[i][j] = {};

  for (int kb = 0; kb < K; kb += 64) {
#pragma unroll
    for (int p = 0; p < 8; ++p) {
      const int idx = tid + p * 256;
      const int m   = idx >> 4;
      const int kc  = (idx & 15) * 4;
      const float4 va = *(const float4*)(A + (size_t)(mb + m) * K + kb + kc);
      unsigned short h0, h1, h2, h3, l0, l1, l2, l3;
      splitbf(va.x, h0, l0);
      splitbf(va.y, h1, l1);
      splitbf(va.z, h2, l2);
      splitbf(va.w, h3, l3);
      ushort4 vh; vh.x = h0; vh.y = h1; vh.z = h2; vh.w = h3;
      ushort4 vl; vl.x = l0; vl.y = l1; vl.z = l2; vl.w = l3;
      *(ushort4*)(&As_hi[m * LDT + kc]) = vh;
      *(ushort4*)(&As_lo[m * LDT + kc]) = vl;
    }
#pragma unroll
    for (int p = 0; p < 8; ++p) {
      const int k4 = (tid >> 7) * 4 + p * 8;
      const int n  = tid & 127;
      unsigned short h[4], l[4];
#pragma unroll
      for (int i = 0; i < 4; ++i) {
        const float bv = B[(size_t)(kb + k4 + i) * N + nb + n];
        splitbf(bv, h[i], l[i]);
      }
      ushort4 vh; vh.x = h[0]; vh.y = h[1]; vh.z = h[2]; vh.w = h[3];
      ushort4 vl; vl.x = l[0]; vl.y = l[1]; vl.z = l[2]; vl.w = l[3];
      *(ushort4*)(&Bs_hi[n * LDT + k4]) = vh;
      *(ushort4*)(&Bs_lo[n * LDT + k4]) = vl;
    }
    __syncthreads();

#pragma unroll
    for (int ks = 0; ks < 2; ++ks) {
      const int koff = ks * 32;
      v16bf Bh[2], Bl[2];
#pragma unroll
      for (int j = 0; j < 2; ++j) {
        const int base = (wn + j * 16 + ln) * LDT + koff + 16 * g;
        Bh[j] = cat8(*(const v8bf*)(&Bs_hi[base]),
                     *(const v8bf*)(&Bs_hi[base + 8]));
        Bl[j] = cat8(*(const v8bf*)(&Bs_lo[base]),
                     *(const v8bf*)(&Bs_lo[base + 8]));
      }
#pragma unroll
      for (int i = 0; i < 4; ++i) {
        const int abase = (wm + i * 16 + ln) * LDT + koff + 16 * g;
        const v16bf Ah = cat8(*(const v8bf*)(&As_hi[abase]),
                              *(const v8bf*)(&As_hi[abase + 8]));
        const v16bf Al = cat8(*(const v8bf*)(&As_lo[abase]),
                              *(const v8bf*)(&As_lo[abase + 8]));
#pragma unroll
        for (int j = 0; j < 2; ++j) {
          acc[i][j] = wmma32(Ah, Bh[j], acc[i][j]);  // hi*hi
          acc[i][j] = wmma32(Ah, Bl[j], acc[i][j]);  // hi*lo
          acc[i][j] = wmma32(Al, Bh[j], acc[i][j]);  // lo*hi
        }
      }
    }
    __syncthreads();
  }

#pragma unroll
  for (int i = 0; i < 4; ++i)
#pragma unroll
    for (int j = 0; j < 2; ++j)
#pragma unroll
      for (int r = 0; r < 8; ++r)
        C[(size_t)(mb + wm + i * 16 + r + 8 * g) * N + nb + wn + j * 16 + ln] =
            acc[i][j][r];
}

// ---------------------------------------------------------------------------
// Pre-split K (row-major) and V (transposed) into bf16 hi/lo for attention.
//   k_hi/k_lo : [B, S, 128]      vT_hi/vT_lo : [B, 128, S]
// ---------------------------------------------------------------------------
__global__ __launch_bounds__(256) void split_kv(
    const float* __restrict__ qkv, unsigned short* __restrict__ k_hi,
    unsigned short* __restrict__ k_lo, unsigned short* __restrict__ vT_hi,
    unsigned short* __restrict__ vT_lo) {
  const int idx = blockIdx.x * 256 + threadIdx.x;  // [0, B*S*128)
  const int d = idx & 127;
  const int s = (idx >> 7) & (SEQ - 1);
  const int b = idx >> 18;
  const size_t row = ((size_t)b * SEQ + s) * QKV_N;
  unsigned short h, l;
  splitbf(qkv[row + EMB + d], h, l);
  k_hi[idx] = h;
  k_lo[idx] = l;
  splitbf(qkv[row + EMB + HEAD_DIM + d], h, l);
  const size_t tix = ((size_t)b * HEAD_DIM + d) * SEQ + s;
  vT_hi[tix] = h;
  vT_lo[tix] = l;
}

// ---------------------------------------------------------------------------
// MQA attention with ALiBi + causal mask, FlashAttention-style online softmax.
// One wave per (batch, head, 16-row q tile); 32-column j blocks; bf16x3 WMMA
// for both Q@K^T and P@V; softmax stats in fp32.
// ---------------------------------------------------------------------------
__global__ __launch_bounds__(32) void mqa_alibi_attn(
    const float* __restrict__ qkv, const unsigned short* __restrict__ k_hi,
    const unsigned short* __restrict__ k_lo,
    const unsigned short* __restrict__ vT_hi,
    const unsigned short* __restrict__ vT_lo, float* __restrict__ attn) {
  const int qt   = blockIdx.x;
  const int h    = blockIdx.y;
  const int b    = blockIdx.z;
  const int lane = threadIdx.x;
  const int ln   = lane & 15;
  const int g    = lane >> 4;
  const int qbase = qt * 16;

  constexpr int LDP = 40;                 // bf16 stride, keeps b128 aligned
  __shared__ unsigned short plds_hi[16 * LDP];
  __shared__ unsigned short plds_lo[16 * LDP];

  const float scaling = 0.08838834764831845f;           // 128^-0.5
  const float slope   = exp2f(-0.25f * (float)(h + 1)); // ALiBi, H=32 pow2

  // Q: split once into bf16x2 A-fragments (4 k-steps of 32)
  const float* qrow =
      qkv + ((size_t)b * SEQ + qbase + ln) * QKV_N + h * HEAD_DIM;
  v16bf Qh[4], Ql[4];
#pragma unroll
  for (int t = 0; t < 4; ++t) {
    v8bf h0, l0, h1, l1;
    split8(qrow + 32 * t + 16 * g, h0, l0);
    split8(qrow + 32 * t + 16 * g + 8, h1, l1);
    Qh[t] = cat8(h0, h1);
    Ql[t] = cat8(l0, l1);
  }

  v8f o[8];
#pragma unroll
  for (int f = 0; f < 8; ++f) o[f] = {};
  float m[8], l[8];
#pragma unroll
  for (int r = 0; r < 8; ++r) { m[r] = -3.0e38f; l[r] = 0.0f; }

  const int nblk = qt / 2 + 1;   // 32-col blocks covering j <= qbase+15
  for (int jb = 0; jb < nblk; ++jb) {
    const int jbase = jb * 32;

    // scores for two 16-col halves
    float val[2][8];
#pragma unroll
    for (int half = 0; half < 2; ++half) {
      const unsigned short* krh =
          k_hi + ((size_t)b * SEQ + jbase + half * 16 + ln) * HEAD_DIM;
      const unsigned short* krl =
          k_lo + ((size_t)b * SEQ + jbase + half * 16 + ln) * HEAD_DIM;
      v8f s = {};
#pragma unroll
      for (int t = 0; t < 4; ++t) {
        const int off = 32 * t + 16 * g;
        const v16bf Kh = cat8(*(const v8bf*)(krh + off),
                              *(const v8bf*)(krh + off + 8));
        const v16bf Kl = cat8(*(const v8bf*)(krl + off),
                              *(const v8bf*)(krl + off + 8));
        s = wmma32(Qh[t], Kh, s);
        s = wmma32(Qh[t], Kl, s);
        s = wmma32(Ql[t], Kh, s);
      }
      const int j = jbase + half * 16 + ln;
#pragma unroll
      for (int r = 0; r < 8; ++r) {
        const int q = qbase + r + 8 * g;
        float v = s[r] * scaling + slope * (float)(j - q);
        if (j > q) v = -3.0e38f;
        val[half][r] = v;
      }
    }

    // online softmax across the 32 columns (rows live in 16-lane groups)
    float p0[8], p1[8], corr[8];
#pragma unroll
    for (int r = 0; r < 8; ++r) {
      float red = fmaxf(val[0][r], val[1][r]);
      red = fmaxf(red, __shfl_xor(red, 1, 32));
      red = fmaxf(red, __shfl_xor(red, 2, 32));
      red = fmaxf(red, __shfl_xor(red, 4, 32));
      red = fmaxf(red, __shfl_xor(red, 8, 32));
      const float mn = fmaxf(m[r], red);
      p0[r] = __expf(val[0][r] - mn);
      p1[r] = __expf(val[1][r] - mn);
      float rs = p0[r] + p1[r];
      rs += __shfl_xor(rs, 1, 32);
      rs += __shfl_xor(rs, 2, 32);
      rs += __shfl_xor(rs, 4, 32);
      rs += __shfl_xor(rs, 8, 32);
      corr[r] = __expf(m[r] - mn);
      l[r]    = l[r] * corr[r] + rs;
      m[r]    = mn;
    }
#pragma unroll
    for (int f = 0; f < 8; ++f)
#pragma unroll
      for (int r = 0; r < 8; ++r) o[f][r] *= corr[r];

    // P (16x32): split to bf16 hi/lo, relay C layout -> A layout via LDS
    __syncthreads();
#pragma unroll
    for (int r = 0; r < 8; ++r) {
      const int row = (r + 8 * g) * LDP;
      unsigned short hh, ll;
      splitbf(p0[r], hh, ll);
      plds_hi[row + ln] = hh;
      plds_lo[row + ln] = ll;
      splitbf(p1[r], hh, ll);
      plds_hi[row + 16 + ln] = hh;
      plds_lo[row + 16 + ln] = ll;
    }
    __syncthreads();
    const int pb = ln * LDP + 16 * g;
    const v16bf Ph = cat8(*(const v8bf*)(&plds_hi[pb]),
                          *(const v8bf*)(&plds_hi[pb + 8]));
    const v16bf Pl = cat8(*(const v8bf*)(&plds_lo[pb]),
                          *(const v8bf*)(&plds_lo[pb + 8]));

    // O += P @ V (one K=32 WMMA step per d-tile)
#pragma unroll
    for (int f = 0; f < 8; ++f) {
      const size_t vo =
          ((size_t)b * HEAD_DIM + f * 16 + ln) * SEQ + jbase + 16 * g;
      const v16bf Vh =
          cat8(*(const v8bf*)(vT_hi + vo), *(const v8bf*)(vT_hi + vo + 8));
      const v16bf Vl =
          cat8(*(const v8bf*)(vT_lo + vo), *(const v8bf*)(vT_lo + vo + 8));
      o[f] = wmma32(Ph, Vh, o[f]);
      o[f] = wmma32(Ph, Vl, o[f]);
      o[f] = wmma32(Pl, Vh, o[f]);
    }
  }

  // normalize + store to attn[B,S,E] (bqhd flattened)
  float inv[8];
#pragma unroll
  for (int r = 0; r < 8; ++r) inv[r] = 1.0f / l[r];
#pragma unroll
  for (int f = 0; f < 8; ++f)
#pragma unroll
    for (int r = 0; r < 8; ++r)
      attn[((size_t)b * SEQ + qbase + r + 8 * g) * EMB + h * HEAD_DIM +
           f * 16 + ln] = o[f][r] * inv[r];
}

// ---------------------------------------------------------------------------
extern "C" void kernel_launch(void* const* d_in, const int* in_sizes, int n_in,
                              void* d_out, int out_size, void* d_ws,
                              size_t ws_size, hipStream_t stream) {
  const float* hs    = (const float*)d_in[0];   // [B,S,E]
  const float* w_qkv = (const float*)d_in[1];   // [E, 4352]
  const float* w_c   = (const float*)d_in[2];   // [E, E]
  float* out = (float*)d_out;                   // [B,S,E]

  float* qkv  = (float*)d_ws;                               // [B*S, 4352]
  float* attn = qkv + (size_t)BATCH * SEQ * QKV_N;          // [B*S, E]
  unsigned short* kv = (unsigned short*)(attn + (size_t)BATCH * SEQ * EMB);
  const size_t NK = (size_t)BATCH * SEQ * HEAD_DIM;
  unsigned short* k_hi  = kv;
  unsigned short* k_lo  = kv + NK;
  unsigned short* vT_hi = kv + 2 * NK;
  unsigned short* vT_lo = kv + 3 * NK;

  // 1) qkv = hidden @ w_qkv   (M=4096, K=4096, N=4352)
  dim3 g1(QKV_N / 128, (BATCH * SEQ) / 128);
  gemm_bf16x3<<<g1, 256, 0, stream>>>(hs, w_qkv, qkv, BATCH * SEQ, QKV_N, EMB);
  // 2a) split K/V to bf16 hi/lo (V transposed)
  split_kv<<<(BATCH * SEQ * HEAD_DIM) / 256, 256, 0, stream>>>(
      qkv, k_hi, k_lo, vT_hi, vT_lo);
  // 2b) attention
  dim3 g2(SEQ / 16, NUM_HEADS, BATCH);
  mqa_alibi_attn<<<g2, 32, 0, stream>>>(qkv, k_hi, k_lo, vT_hi, vT_lo, attn);
  // 3) out = attn @ w_c       (M=4096, K=4096, N=4096)
  dim3 g3(EMB / 128, (BATCH * SEQ) / 128);
  gemm_bf16x3<<<g3, 256, 0, stream>>>(attn, w_c, out, BATCH * SEQ, EMB, EMB);
}
